// EGBLSTM_42649025249295
// MI455X (gfx1250) — matrix operand order repaired and checked by reading, not compiled, not run on hardware
//
#include <hip/hip_runtime.h>

// ---------------- problem constants ----------------
#define BB 2048
#define SS 256
#define HH 64
#define SDI 4
#define MDI 2
#define BT 16          // batches per workgroup (WMMA M)
#define NT 128         // threads per workgroup (4 waves)
#define KG 160         // padded K for gates ([c(64), st/60(4), h(64)] -> 132 -> 160)
#define KF 96          // padded K for heads (68/70 -> 96)
#define NG 256         // gate outputs (4*H)

typedef __attribute__((ext_vector_type(16))) _Float16 v16h;
typedef __attribute__((ext_vector_type(8)))  _Float16 h8;
typedef __attribute__((ext_vector_type(8)))  float    v8f;

union V16U { v16h v; h8 h[2]; };

// ---------------- LDS layout (bytes) ----------------
constexpr int O_WG   = 0;                         // [256][160] f16  gate weights (N-major)
constexpr int O_WF0  = O_WG  + NG*KG*2;           // [64][96] f16
constexpr int O_WS0  = O_WF0 + HH*KF*2;           // [64][96] f16
constexpr int O_WH0  = O_WS0 + HH*KF*2;           // [64][96] f16
constexpr int O_ZA   = O_WH0 + HH*KF*2;           // [16][160] f16  gate A operand
constexpr int O_FA   = O_ZA  + BT*KG*2;           // [16][96]  f16  feat A operand
constexpr int O_F2A  = O_FA  + BT*KF*2;           // [16][96]  f16  feat2 A operand
constexpr int O_HF0  = O_F2A + BT*KF*2;           // [16][64]  f32  tanh(feat@Wf0)
constexpr int O_HS0  = O_HF0 + BT*HH*4;           // [16][64]  f32
constexpr int O_HH0  = O_HS0 + BT*HH*4;           // [16][64]  f32
constexpr int O_DOUT = O_HH0 + BT*HH*4;           // [16][8]   f32  delta/pvec/hvec
constexpr int O_GB   = O_DOUT+ BT*8*4;            // [256] f32 gate bias (b_ih+b_hh)
constexpr int O_BF0  = O_GB  + NG*4;              // [64] f32
constexpr int O_BS0  = O_BF0 + HH*4;              // [64] f32
constexpr int O_BH0  = O_BS0 + HH*4;              // [64] f32
constexpr int O_WF1  = O_BH0 + HH*4;              // [2][64] f32 (TRANSPOSED: [o][k])
constexpr int O_WS1  = O_WF1 + 128*4;             // [4][64] f32 (TRANSPOSED)
constexpr int O_WH1  = O_WS1 + 256*4;             // [2][64] f32 (TRANSPOSED)
constexpr int O_SM   = O_WH1 + 128*4;             // 64 f32: F(0..15) G(16..23) Hm(24..31) Q(32..47) R(48..51) bf1(52..53) bs1(54..57) bh1(58..59)
constexpr int SMEM_TOTAL = O_SM + 64*4;

// hardware tanh (gfx1250 TRANS op); v_nop satisfies the TRANS RAW hazard
// (ISA 7.4: 1 independent op or V_NOP before the trans result may be consumed)
__device__ __forceinline__ float tanh_hw(float x) {
  float y;
  asm("v_tanh_f32 %0, %1\n\tv_nop" : "=v"(y) : "v"(x));
  return y;
}
__device__ __forceinline__ float sigf(float x) { return fmaf(0.5f, tanh_hw(0.5f * x), 0.5f); }

// A operand (16x32 f16): lane<16 rows M=lane K[0..7]+K[16..23]; lane>=16 K[8..15]+K[24..31]
__device__ __forceinline__ v16h load_a(const _Float16* row, int kt, int koff) {
  V16U u;
  u.h[0] = *(const h8*)(row + 32*kt + koff);
  u.h[1] = *(const h8*)(row + 32*kt + 16 + koff);
  return u.v;
}
// B operand (32x16 f16): lane<16 col=lane K[0..7]+K[16..23]; lane>=16 col=lane-16 K[8..15]+K[24..31]
__device__ __forceinline__ v16h load_b(const _Float16* wrow, int kt, int kbase) {
  V16U u;
  u.h[0] = *(const h8*)(wrow + 32*kt + kbase / 2);
  u.h[1] = *(const h8*)(wrow + 32*kt + kbase / 2 + 16);
  return u.v;
}
__device__ __forceinline__ v8f wmma16(v16h a, v16h b, v8f c) {
  return __builtin_amdgcn_wmma_f32_16x16x32_f16(false, a, false, b, (short)0, c, false, false);
}

extern __shared__ char smem[];

__global__ __launch_bounds__(NT, 1)
void egblstm_scan(const float* __restrict__ x, const float* __restrict__ target,
                  const float* __restrict__ c0, const float* __restrict__ h0,
                  const float* __restrict__ lc0,
                  const float* __restrict__ F, const float* __restrict__ G,
                  const float* __restrict__ Hm, const float* __restrict__ Q,
                  const float* __restrict__ R,
                  const float* __restrict__ W_ih, const float* __restrict__ W_hh,
                  const float* __restrict__ b_ih, const float* __restrict__ b_hh,
                  const float* __restrict__ Wf0, const float* __restrict__ bf0,
                  const float* __restrict__ Wf1, const float* __restrict__ bf1,
                  const float* __restrict__ Ws0, const float* __restrict__ bs0,
                  const float* __restrict__ Ws1, const float* __restrict__ bs1,
                  const float* __restrict__ Wh0, const float* __restrict__ bh0,
                  const float* __restrict__ Wh1, const float* __restrict__ bh1,
                  float* __restrict__ out) {
  _Float16* Wg_l   = (_Float16*)(smem + O_WG);
  _Float16* Wf0_l  = (_Float16*)(smem + O_WF0);
  _Float16* Ws0_l  = (_Float16*)(smem + O_WS0);
  _Float16* Wh0_l  = (_Float16*)(smem + O_WH0);
  _Float16* zA     = (_Float16*)(smem + O_ZA);
  _Float16* featA  = (_Float16*)(smem + O_FA);
  _Float16* feat2A = (_Float16*)(smem + O_F2A);
  float* hf0_l  = (float*)(smem + O_HF0);
  float* hs0_l  = (float*)(smem + O_HS0);
  float* hh0_l  = (float*)(smem + O_HH0);
  float* dout_l = (float*)(smem + O_DOUT);
  float* gb_l   = (float*)(smem + O_GB);
  float* bf0_l  = (float*)(smem + O_BF0);
  float* bs0_l  = (float*)(smem + O_BS0);
  float* bh0_l  = (float*)(smem + O_BH0);
  float* Wf1_l  = (float*)(smem + O_WF1);
  float* Ws1_l  = (float*)(smem + O_WS1);
  float* Wh1_l  = (float*)(smem + O_WH1);
  float* sm     = (float*)(smem + O_SM);

  const int tt    = threadIdx.x;
  const int wave  = tt >> 5;
  const int lane  = tt & 31;
  const int colx  = lane & 15;            // A: batch row; B: column within tile
  const int hid   = wave * 16 + colx;     // hidden column owned by this lane
  const int mbase = (lane < 16) ? 0 : 8;  // C/D row base
  const int koff  = (lane < 16) ? 0 : 8;  // A K sub-offset
  const int kbase = (lane < 16) ? 0 : 16; // B K sub-offset
  const int bg    = blockIdx.x * BT;      // global batch base

  // ---------------- init: weights -> LDS (f32 -> f16) ----------------
  for (int idx = tt; idx < NG * KG; idx += NT) {
    int n = idx / KG, k = idx % KG;
    float v = 0.0f;
    if (k < 68) v = W_ih[k * NG + n];
    else if (k < 132) v = W_hh[(k - 68) * NG + n];
    Wg_l[idx] = (_Float16)v;
  }
  for (int idx = tt; idx < HH * KF; idx += NT) {
    int n = idx / KF, k = idx % KF;
    Wf0_l[idx] = (_Float16)((k < 68) ? Wf0[k * HH + n] : 0.0f);
    Ws0_l[idx] = (_Float16)((k < 68) ? Ws0[k * HH + n] : 0.0f);
    Wh0_l[idx] = (_Float16)((k < 70) ? Wh0[k * HH + n] : 0.0f);
  }
  for (int n = tt; n < NG; n += NT) gb_l[n] = b_ih[n] + b_hh[n];
  for (int n = tt; n < HH; n += NT) { bf0_l[n] = bf0[n]; bs0_l[n] = bs0[n]; bh0_l[n] = bh0[n]; }
  // small projection weights TRANSPOSED: W*_l[o*64 + k] = W*[k*no + o]
  for (int i = tt; i < 128; i += NT) {
    int o = i >> 6, k = i & 63;
    Wf1_l[i] = Wf1[k * 2 + o];
    Wh1_l[i] = Wh1[k * 2 + o];
  }
  for (int i = tt; i < 256; i += NT) {
    int o = i >> 6, k = i & 63;
    Ws1_l[i] = Ws1[k * 4 + o];
  }
  if (tt < 16) sm[tt] = F[tt];
  if (tt < 8)  sm[16 + tt] = G[tt];
  if (tt < 8)  sm[24 + tt] = Hm[tt];
  if (tt < 16) sm[32 + tt] = Q[tt];
  if (tt < 4)  sm[48 + tt] = R[tt];
  if (tt < 2)  sm[52 + tt] = bf1[tt];
  if (tt < 4)  sm[54 + tt] = bs1[tt];
  if (tt < 2)  sm[58 + tt] = bh1[tt];

  // ---------------- init: activation operands ----------------
  for (int idx = tt; idx < BT * KG; idx += NT) {
    int b = idx / KG, k = idx % KG;
    size_t gb = (size_t)(bg + b);
    float v = 0.0f;
    if (k < 64)       v = c0[gb * HH + k];
    else if (k < 68)  v = target[(gb * SS) * SDI + (k - 64)] * (1.0f / 60.0f);
    else if (k < 132) v = h0[gb * HH + (k - 68)];
    zA[idx] = (_Float16)v;
  }
  for (int idx = tt; idx < BT * KF; idx += NT) {
    int b = idx / KF, k = idx % KF;
    float v = 0.0f;
    if (k >= 64 && k < 68)
      v = target[((size_t)(bg + b) * SS) * SDI + (k - 64)] * (1.0f / 60.0f);
    _Float16 hv = (_Float16)v;
    featA[idx] = hv;
    feat2A[idx] = hv;
  }

  // ---------------- persistent per-lane state ----------------
  float lcr[8];
#pragma unroll
  for (int r = 0; r < 8; ++r)
    lcr[r] = lc0[(size_t)(bg + mbase + r) * HH + hid];

  float stx[4], Pm[16];
#pragma unroll
  for (int i = 0; i < 4; ++i)
    stx[i] = (tt < BT) ? target[((size_t)(bg + tt) * SS) * SDI + i] : 0.0f;
#pragma unroll
  for (int i = 0; i < 16; ++i) Pm[i] = ((i & 3) == (i >> 2)) ? 1.0f : 0.0f;

  __syncthreads();

  // ---------------- hoist ALL loop-invariant B fragments + biases to VGPRs ----
  // 20 gate frags + 9 head frags = 29 x v16h (1 WG/WGP, 1 wave/SIMD -> large
  // register budget is the right trade; removes all per-step weight ds traffic)
  v16h bgw[20];
#pragma unroll
  for (int g = 0; g < 4; ++g)
#pragma unroll
    for (int kt = 0; kt < 5; ++kt)
      bgw[g * 5 + kt] = load_b(Wg_l + (g * HH + hid) * KG, kt, kbase);
  v16h bfw[3], bsw[3], bhw[3];
#pragma unroll
  for (int kt = 0; kt < 3; ++kt) {
    bfw[kt] = load_b(Wf0_l + hid * KF, kt, kbase);
    bsw[kt] = load_b(Ws0_l + hid * KF, kt, kbase);
    bhw[kt] = load_b(Wh0_l + hid * KF, kt, kbase);
  }
  float gbr[4];
#pragma unroll
  for (int g = 0; g < 4; ++g) gbr[g] = gb_l[g * HH + hid];
  const float bf0r = bf0_l[hid], bs0r = bs0_l[hid], bh0r = bh0_l[hid];

  const _Float16* zrow  = zA + colx * KG;
  const _Float16* frow  = featA + colx * KF;
  const _Float16* f2row = feat2A + colx * KF;

  // ---------------- sequential scan ----------------
#pragma unroll 1
  for (int t = 0; t < SS; ++t) {
    // ---- Phase A: gates = z @ Wg (first WMMA uses inline C=0; bias folded later)
    v8f acc[4];
    {
      const v8f z8 = {};
      v16h a = load_a(zrow, 0, koff);
#pragma unroll
      for (int g = 0; g < 4; ++g) acc[g] = wmma16(a, bgw[g * 5], z8);
#pragma unroll
      for (int kt = 1; kt < 5; ++kt) {
        a = load_a(zrow, kt, koff);
#pragma unroll
        for (int g = 0; g < 4; ++g) acc[g] = wmma16(a, bgw[g * 5 + kt], acc[g]);
      }
    }
    __syncthreads();

    // ---- Phase B: LSTM pointwise in D regs; publish h (=cu) as f16 operands
#pragma unroll
    for (int r = 0; r < 8; ++r) {
      float ig = sigf(acc[0][r] + gbr[0]);
      float fg = sigf(acc[1][r] + gbr[1]);
      float gg = tanh_hw(acc[2][r] + gbr[2]);
      float og = sigf(acc[3][r] + gbr[3]);
      lcr[r] = fg * lcr[r] + ig * gg;
      float hv = og * tanh_hw(lcr[r]);
      _Float16 h16 = (_Float16)hv;
      int mb = mbase + r;
      zA[mb * KG + hid] = h16;            // c slot for next step (c_next = cu = h)
      zA[mb * KG + 68 + hid] = h16;       // h slot for next step
      featA[mb * KF + hid] = h16;         // cu for feat
      feat2A[mb * KF + hid] = h16;        // cu for feat2
    }
    if (tt < BT * MDI) {
      int b = tt >> 1, md = tt & 1;
      const float* xp = x + ((size_t)(bg + b) * SS + t) * MDI + md;
      feat2A[b * KF + 68 + md] = (_Float16)(xp[0] * (1.0f / 60.0f));
      if (t + 1 < SS) __builtin_prefetch(xp + MDI, 0, 1);  // global_prefetch_b8
    }
    __syncthreads();

    // ---- Phase C: three heads, one 16-col tile per wave (weights in regs)
    v8f df, dsv, dh;
    {
      const v8f z8 = {};
      v16h a1 = load_a(frow, 0, koff);
      v16h a2 = load_a(f2row, 0, koff);
      df  = wmma16(a1, bfw[0], z8);
      dsv = wmma16(a1, bsw[0], z8);
      dh  = wmma16(a2, bhw[0], z8);
#pragma unroll
      for (int kt = 1; kt < 3; ++kt) {
        a1 = load_a(frow, kt, koff);
        a2 = load_a(f2row, kt, koff);
        df  = wmma16(a1, bfw[kt], df);
        dsv = wmma16(a1, bsw[kt], dsv);
        dh  = wmma16(a2, bhw[kt], dh);
      }
    }
#pragma unroll
    for (int r = 0; r < 8; ++r) {
      int mb = mbase + r;
      hf0_l[mb * HH + hid] = tanh_hw(df[r] + bf0r);
      hs0_l[mb * HH + hid] = tanh_hw(dsv[r] + bs0r);
      hh0_l[mb * HH + hid] = tanh_hw(dh[r] + bh0r);
    }
    __syncthreads();

    // ---- Phase D: 64 -> {2,4,2} projections, float4-vectorized dot per thread
    {
      int b = tt >> 3, o = tt & 7;
      const float4* hp4;
      const float4* wp4;
      float accd;
      if (o < 2) {
        accd = sm[52 + o];
        hp4 = (const float4*)(hf0_l + b * HH);
        wp4 = (const float4*)(Wf1_l + o * HH);
      } else if (o < 6) {
        int oo = o - 2;
        accd = sm[54 + oo];
        hp4 = (const float4*)(hs0_l + b * HH);
        wp4 = (const float4*)(Ws1_l + oo * HH);
      } else {
        int oo = o - 6;
        accd = sm[58 + oo];
        hp4 = (const float4*)(hh0_l + b * HH);
        wp4 = (const float4*)(Wh1_l + oo * HH);
      }
#pragma unroll
      for (int k = 0; k < 16; ++k) {
        float4 hv = hp4[k], wv = wp4[k];
        accd = fmaf(hv.x, wv.x, fmaf(hv.y, wv.y, fmaf(hv.z, wv.z, fmaf(hv.w, wv.w, accd))));
      }
      dout_l[b * 8 + o] = accd;
    }
    __syncthreads();

    // ---- Phase E: per-batch 4x4/2x2 Kalman update (lanes 0..15 of wave 0)
    if (tt < BT) {
      const int b = tt;
      float de0 = dout_l[b * 8 + 0], de1 = dout_l[b * 8 + 1];
      float pv[4] = { dout_l[b * 8 + 2], dout_l[b * 8 + 3],
                      dout_l[b * 8 + 4], dout_l[b * 8 + 5] };
      float hv0 = dout_l[b * 8 + 6], hv1 = dout_l[b * 8 + 7];

      float sp[4], FP[16], Pp[16];
#pragma unroll
      for (int i = 0; i < 4; ++i) {
        float s = sm[16 + i * 2 + 0] * de0 + sm[16 + i * 2 + 1] * de1;
#pragma unroll
        for (int j = 0; j < 4; ++j) s += sm[i * 4 + j] * stx[j];
        sp[i] = s;
      }
#pragma unroll
      for (int i = 0; i < 4; ++i)
#pragma unroll
        for (int j = 0; j < 4; ++j) {
          float s = 0.0f;
#pragma unroll
          for (int k = 0; k < 4; ++k) s += sm[i * 4 + k] * Pm[k * 4 + j];
          FP[i * 4 + j] = s;
        }
#pragma unroll
      for (int i = 0; i < 4; ++i)
#pragma unroll
        for (int j = 0; j < 4; ++j) {
          float s = 0.0f;
#pragma unroll
          for (int k = 0; k < 4; ++k) s += FP[i * 4 + k] * sm[j * 4 + k];
          Pp[i * 4 + j] = s + pv[i] * pv[j] + sm[32 + i * 4 + j];
        }
      const float* xp = x + ((size_t)(bg + b) * SS + t) * MDI;
      float in0 = xp[0], in1 = xp[1];
#pragma unroll
      for (int j = 0; j < 4; ++j) { in0 -= sm[24 + j] * sp[j]; in1 -= sm[28 + j] * sp[j]; }
      float HP[8];
#pragma unroll
      for (int m = 0; m < 2; ++m)
#pragma unroll
        for (int j = 0; j < 4; ++j) {
          float s = 0.0f;
#pragma unroll
          for (int k = 0; k < 4; ++k) s += sm[24 + m * 4 + k] * Pp[k * 4 + j];
          HP[m * 4 + j] = s;
        }
      float S2[4];
#pragma unroll
      for (int m = 0; m < 2; ++m)
#pragma unroll
        for (int n = 0; n < 2; ++n) {
          float s = 0.0f;
#pragma unroll
          for (int j = 0; j < 4; ++j) s += HP[m * 4 + j] * sm[24 + n * 4 + j];
          float hm = (m == 0) ? hv0 : hv1, hn = (n == 0) ? hv0 : hv1;
          S2[m * 2 + n] = s + hm * hn + sm[48 + m * 2 + n];
        }
      float rdet = 1.0f / (S2[0] * S2[3] - S2[1] * S2[2]);
      float iS00 = S2[3] * rdet, iS01 = -S2[1] * rdet;
      float iS10 = -S2[2] * rdet, iS11 = S2[0] * rdet;
      float Kk[8], KS[8], su[4];
#pragma unroll
      for (int i = 0; i < 4; ++i) {
        float p0 = 0.0f, p1 = 0.0f;
#pragma unroll
        for (int j = 0; j < 4; ++j) { p0 += Pp[i * 4 + j] * sm[24 + j]; p1 += Pp[i * 4 + j] * sm[28 + j]; }
        Kk[i * 2 + 0] = p0 * iS00 + p1 * iS10;
        Kk[i * 2 + 1] = p0 * iS01 + p1 * iS11;
        su[i] = sp[i] + Kk[i * 2 + 0] * in0 + Kk[i * 2 + 1] * in1;
        KS[i * 2 + 0] = Kk[i * 2 + 0] * S2[0] + Kk[i * 2 + 1] * S2[2];
        KS[i * 2 + 1] = Kk[i * 2 + 0] * S2[1] + Kk[i * 2 + 1] * S2[3];
      }
#pragma unroll
      for (int i = 0; i < 4; ++i)
#pragma unroll
        for (int j = 0; j < 4; ++j)
          Pm[i * 4 + j] = Pp[i * 4 + j] - KS[i * 2 + 0] * Kk[j * 2 + 0] - KS[i * 2 + 1] * Kk[j * 2 + 1];

      float* op = out + ((size_t)(bg + b) * SS + t) * SDI;
#pragma unroll
      for (int i = 0; i < 4; ++i) {
        stx[i] = su[i];
        op[i] = su[i];
        _Float16 sv = (_Float16)(su[i] * (1.0f / 60.0f));
        zA[b * KG + 64 + i] = sv;       // st carry for next step's gates
        featA[b * KF + 64 + i] = sv;    // st carry for next step's heads
        feat2A[b * KF + 64 + i] = sv;
      }
    }
    __syncthreads();  // E writes -> next-step A/C reads
  }
}

extern "C" void kernel_launch(void* const* d_in, const int* in_sizes, int n_in,
                              void* d_out, int out_size, void* d_ws, size_t ws_size,
                              hipStream_t stream) {
  (void)in_sizes; (void)n_in; (void)out_size; (void)d_ws; (void)ws_size;
  hipFuncSetAttribute((const void*)egblstm_scan,
                      hipFuncAttributeMaxDynamicSharedMemorySize, SMEM_TOTAL);
  egblstm_scan<<<BB / BT, NT, SMEM_TOTAL, stream>>>(
      (const float*)d_in[0],  (const float*)d_in[1],  (const float*)d_in[2],
      (const float*)d_in[3],  (const float*)d_in[4],  (const float*)d_in[5],
      (const float*)d_in[6],  (const float*)d_in[7],  (const float*)d_in[8],
      (const float*)d_in[9],  (const float*)d_in[10], (const float*)d_in[11],
      (const float*)d_in[12], (const float*)d_in[13], (const float*)d_in[14],
      (const float*)d_in[15], (const float*)d_in[16], (const float*)d_in[17],
      (const float*)d_in[18], (const float*)d_in[19], (const float*)d_in[20],
      (const float*)d_in[21], (const float*)d_in[22], (const float*)d_in[23],
      (const float*)d_in[24], (const float*)d_in[25], (float*)d_out);
}
